// opt_loss_54150947668663
// MI455X (gfx1250) — compile-verified, matrix operand order checked by reading, not run on hardware
//
#include <hip/hip_runtime.h>

typedef __attribute__((ext_vector_type(16))) _Float16 v16h;
typedef __attribute__((ext_vector_type(8)))  _Float16 v8h;
typedef __attribute__((ext_vector_type(8)))  float    v8f;

// ---------------- problem constants ----------------
#define N1F     16384.0f
#define NSTEP   511            // N2 - 1
#define DTF     (1.0f/512.0f)
#define RRATE   0.05f
#define K0c     0.2f
#define K1c     0.25f
#define S0c     0.3f
#define S1c     0.35f
#define INVP0   1.0f
#define INVP1   (1.0f/1.2f)
#define RRc     0.5f
#define Bc      1.0f
#define Qc      1.0f
#define UAc     1.0f
#define UBc     1.0f
#define UA_G    0.5f           // UA/GAM with GAM=2
#define Lc      0.1f
#define OMc     0.5f
#define DDc     0.05f
#define PIc     1.0f
#define XI0     1.0f
#define XI1     1.0f
#define LAMc    10.0f

// ---------------- workspace layout (floats) ----------------
enum {
  OFF_V1B = 0,
  OFF_V2B = OFF_V1B + NSTEP,
  OFF_Z1  = OFF_V2B + NSTEP,
  OFF_Z2  = OFF_Z1  + NSTEP,
  OFF_CM  = OFF_Z2  + NSTEP,
  OFF_PS1 = OFF_CM  + NSTEP,
  OFF_PS2 = OFF_PS1 + NSTEP,
  OFF_TERM= OFF_PS2 + NSTEP,
  OFF_P10 = OFF_TERM + 1,
  OFF_P20 = OFF_P10 + 1,
  WS_FLOATS = OFF_P20 + 1
};

struct Mlp { const float *W0, *b0, *W1, *b1, *W2, *b2; };

__device__ __forceinline__ float clamp01(float v){ return fminf(fmaxf(v, 0.0f), 1.0f); }

// ---------------- kernel 0: zero the accumulation region ----------------
__global__ void zero_kernel(float* ws){
  int i = blockIdx.x*blockDim.x + threadIdx.x;
  int n = OFF_P10 - OFF_PS1;           // psum1, psum2, term_sum
  if (i < n) ws[OFF_PS1 + i] = 0.0f;
}

// ---------------- kernel 1: scalar track precompute (1 block, 64 threads) ----------------
__device__ __forceinline__ float scalar_mlp(const Mlp& m, float x, float* h1, float* h2, float* outsh){
  const int tid = threadIdx.x;                 // 64 threads
  h1[tid] = fmaxf(0.0f, fmaf(x, m.W0[tid], m.b0[tid]));
  __syncthreads();
  float acc = m.b1[tid];
  for (int k = 0; k < 64; ++k) acc = fmaf(h1[k], m.W1[k*64 + tid], acc);
  h2[tid] = fmaxf(acc, 0.0f);
  __syncthreads();
  if (tid == 0){
    float o = m.b2[0];
    for (int k = 0; k < 64; ++k) o = fmaf(h2[k], m.W2[k], o);
    *outsh = o;
  }
  __syncthreads();
  float r = *outsh;
  __syncthreads();
  return r;
}

__global__ void precompute_kernel(Mlp n1, Mlp n2, Mlp n5, Mlp n6, float* ws){
  __shared__ float h1[64], h2[64], outsh;
  const int tid = threadIdx.x;
  float p10 = scalar_mlp(n5, XI0, h1, h2, &outsh);
  float p20 = scalar_mlp(n6, XI1, h1, h2, &outsh);
  if (tid == 0){ ws[OFF_P10] = p10; ws[OFF_P20] = p20; }
  float z1 = XI0, z2 = XI1;
  for (int j = 0; j < NSTEP; ++j){
    float t = (float)j * DTF;
    float v1b = clamp01(scalar_mlp(n1, t, h1, h2, &outsh));
    float v2b = clamp01(scalar_mlp(n2, t, h1, h2, &outsh));
    float cm  = OMc*(K0c - DDc)*v1b + OMc*(K1c - DDc)*v2b;
    if (tid == 0){
      ws[OFF_V1B + j] = v1b;  ws[OFF_V2B + j] = v2b;  ws[OFF_CM + j] = cm;
      ws[OFF_Z1 + j] = z1;    ws[OFF_Z2 + j] = z2;
      z1 = z1 + (RRATE*z1 + Lc - v1b*K0c + cm*PIc)*DTF;
      z2 = z2 + (RRATE*z2 + Lc - v2b*K1c + cm*PIc)*DTF;
    }
  }
}

// ---------------- batched 4->64->64->1 MLP, layer2 via WMMA ----------------
// WMMA phase is grouped: 8 independent K-chunk-0 WMMAs, then 8 K-chunk-1 WMMAs
// (each >=7 instructions away from its RAW producer), then all relu/stores —
// no v_nop hazard fill, back-to-back XDL issue.
__device__ __forceinline__ float mlp64(
    float i0, float i1, float i2, float i3,
    const float* W0s, const float* b0s, const float* W2s, float b2r,
    const v16h (&B)[2][4], const float (&cb)[4],
    _Float16 (*h1)[80], _Float16 (*h2)[80],
    int lane, int nsel, int khalf)
{
  // layer 1: per-lane VALU (K=4), write f16 activations to wave-private LDS tile
  #pragma unroll 8
  for (int jj = 0; jj < 64; ++jj){
    float a = b0s[jj];
    a = fmaf(i0, W0s[jj],       a);
    a = fmaf(i1, W0s[64  + jj], a);
    a = fmaf(i2, W0s[128 + jj], a);
    a = fmaf(i3, W0s[192 + jj], a);
    h1[lane][jj] = (_Float16)fmaxf(a, 0.0f);
  }
  __syncthreads();

  // load A fragments for both 16-row tiles up front (one ds batch)
  v16h A[2][2];                       // [mt][kchunk]
  #pragma unroll
  for (int mt = 0; mt < 2; ++mt){
    const int row = mt*16 + nsel;
    const v8h* rp = (const v8h*)&h1[row][0];
    v8h lo0 = rp[khalf],     hi0 = rp[2 + khalf];   // K {0..7|8..15}, {16..23|24..31}
    v8h lo1 = rp[4 + khalf], hi1 = rp[6 + khalf];   // +32
    #pragma unroll
    for (int h = 0; h < 8; ++h){
      A[mt][0][h] = lo0[h]; A[mt][0][h+8] = hi0[h];
      A[mt][1][h] = lo1[h]; A[mt][1][h+8] = hi1[h];
    }
  }

  // layer 2: 32x64 @ 64x64 via 16 x v_wmma_f32_16x16x32_f16
  v8f c[2][4];
  #pragma unroll
  for (int mt = 0; mt < 2; ++mt)
    #pragma unroll
    for (int nt = 0; nt < 4; ++nt)
      #pragma unroll
      for (int v = 0; v < 8; ++v) c[mt][nt][v] = cb[nt];   // bias-initialized

  #pragma unroll
  for (int mt = 0; mt < 2; ++mt)
    #pragma unroll
    for (int nt = 0; nt < 4; ++nt)
      c[mt][nt] = __builtin_amdgcn_wmma_f32_16x16x32_f16(
          false, A[mt][0], false, B[0][nt], (short)0, c[mt][nt], false, false);
  #pragma unroll
  for (int mt = 0; mt < 2; ++mt)
    #pragma unroll
    for (int nt = 0; nt < 4; ++nt)
      c[mt][nt] = __builtin_amdgcn_wmma_f32_16x16x32_f16(
          false, A[mt][1], false, B[1][nt], (short)0, c[mt][nt], false, false);

  // relu + stage h2 back to LDS
  #pragma unroll
  for (int mt = 0; mt < 2; ++mt)
    #pragma unroll
    for (int nt = 0; nt < 4; ++nt)
      #pragma unroll
      for (int v = 0; v < 8; ++v)
        h2[mt*16 + (khalf<<3) + v][nt*16 + nsel] = (_Float16)fmaxf(c[mt][nt][v], 0.0f);
  __syncthreads();

  // layer 3: per-lane dot of own row with W2 (64x1)
  const v8h* rp2 = (const v8h*)&h2[lane][0];
  float acc = b2r;
  #pragma unroll
  for (int g = 0; g < 8; ++g){
    v8h seg = rp2[g];
    #pragma unroll
    for (int h = 0; h < 8; ++h) acc = fmaf((float)seg[h], W2s[g*8 + h], acc);
  }
  __syncthreads();
  return acc;
}

// ---------------- kernel 2: persistent SDE rollout (128 blocks x 128 threads) ----------------
__global__ __launch_bounds__(128) void sde_kernel(const float* __restrict__ dW, Mlp n3, Mlp n4, float* ws){
  __shared__ __align__(16) _Float16 h1buf[4][32][80];
  __shared__ __align__(16) _Float16 h2buf[4][32][80];
  __shared__ float W0a[256], b0a[64], W2a[64];
  __shared__ float W0b[256], b0b[64], W2b[64];

  const int tid   = threadIdx.x;
  const int lane  = tid & 31, wv = tid >> 5;
  const int nsel  = lane & 15, khalf = lane >> 4;

  for (int i = tid; i < 256; i += 128){ W0a[i] = n3.W0[i]; W0b[i] = n4.W0[i]; }
  if (tid < 64){
    b0a[tid] = n3.b0[tid]; b0b[tid] = n4.b0[tid];
    W2a[tid] = n3.W2[tid]; W2b[tid] = n4.W2[tid];
  }
  __syncthreads();
  const float b2a = n3.b2[0], b2b = n4.b2[0];

  // persistent B fragments (64x64 hidden weights) + bias fragments
  v16h B3[2][4], B4[2][4];
  float cb3[4], cb4[4];
  #pragma unroll
  for (int nt = 0; nt < 4; ++nt){ cb3[nt] = n3.b1[nt*16 + nsel]; cb4[nt] = n4.b1[nt*16 + nsel]; }
  #pragma unroll
  for (int kc = 0; kc < 2; ++kc){
    #pragma unroll
    for (int nt = 0; nt < 4; ++nt){
      v16h a3, a4;
      #pragma unroll
      for (int h = 0; h < 16; ++h){
        int K   = (h & 7) + ((h >> 3) << 4) + (khalf << 3) + (kc << 5);
        int idx = K*64 + nt*16 + nsel;
        a3[h] = (_Float16)n3.W1[idx];
        a4[h] = (_Float16)n4.W1[idx];
      }
      B3[kc][nt] = a3; B4[kc][nt] = a4;
    }
  }

  const int path = blockIdx.x*128 + tid;
  const float2* dwp = (const float2*)(dW + (size_t)path * 1024);  // N2*2 floats per path
  float x1 = XI0, x2 = XI1;
  float p1 = ws[OFF_P10], p2 = ws[OFF_P20];
  _Float16 (*h1)[80] = h1buf[wv];
  _Float16 (*h2)[80] = h2buf[wv];

  for (int j = 0; j < NSTEP; ++j){
    float t   = (float)j * DTF;
    float vb1 = ws[OFF_V1B + j], vb2 = ws[OFF_V2B + j];
    float zz1 = ws[OFF_Z1  + j], zz2 = ws[OFF_Z2  + j];
    float cm  = ws[OFF_CM  + j];
    float2 dw = dwp[j];
    __builtin_prefetch(&dwp[j + 16], 0, 0);

    float eta1 = mlp64(t, x1, zz1, p1, W0a, b0a, W2a, b2a, B3, cb3, h1, h2, lane, nsel, khalf);
    float eta2 = mlp64(t, x2, zz2, p2, W0b, b0b, W2b, b2b, B4, cb4, h1, h2, lane, nsel, khalf);

    float v1 = clamp01(fmaf(K0c, p1, S0c*eta1)*INVP0 + RRc*vb1);
    float v2 = clamp01(fmaf(K1c, p2, S1c*eta2)*INVP1 + RRc*vb2);

    float y1 = (x1 >= 0.0f) ? fmaf(x1, UA_G, UBc) : UBc;
    float y2 = (x2 >= 0.0f) ? fmaf(x2, UA_G, UBc) : UBc;
    float p1n = p1 + (-RRATE*p1 - (x1 - Bc)*Qc + UAc/(y1*y1))*DTF + eta1*dw.x;
    float p2n = p2 + (-RRATE*p2 - (x2 - Bc)*Qc + UAc/(y2*y2))*DTF + eta2*dw.y;
    float x1n = x1 + (RRATE*x1 + Lc - K0c*v1 + cm*PIc)*DTF + S0c*(1.0f - v1)*dw.x;
    float x2n = x2 + (RRATE*x2 + Lc - K1c*v2 + cm*PIc)*DTF + S1c*(1.0f - v2)*dw.y;
    x1 = x1n; x2 = x2n; p1 = p1n; p2 = p2n;

    // penalty partial sums: wave32 reduce + one atomic per wave per step
    float s1 = v1, s2 = v2;
    #pragma unroll
    for (int off = 16; off > 0; off >>= 1){
      s1 += __shfl_xor(s1, off, 32);
      s2 += __shfl_xor(s2, off, 32);
    }
    if (lane == 0){
      atomicAdd(&ws[OFF_PS1 + j], s1);
      atomicAdd(&ws[OFF_PS2 + j], s2);
    }
  }

  // terminal contribution
  float y1 = (x1 >= 0.0f) ? fmaf(x1, UA_G, UBc) : UBc;
  float y2 = (x2 >= 0.0f) ? fmaf(x2, UA_G, UBc) : UBc;
  float t1 = p1 + UAc/(y1*y1) - (x1 - Bc)*Qc;
  float t2 = p2 + UAc/(y2*y2) - (x2 - Bc)*Qc;
  float ss = t1*t1 + t2*t2;
  #pragma unroll
  for (int off = 16; off > 0; off >>= 1) ss += __shfl_xor(ss, off, 32);
  if (lane == 0) atomicAdd(&ws[OFF_TERM], ss);
}

// ---------------- kernel 3: finalize loss & pen ----------------
__global__ void finalize_kernel(const float* ws, float* out){
  if (threadIdx.x == 0 && blockIdx.x == 0){
    float pen = 0.0f;
    for (int j = 0; j < NSTEP; ++j){
      float m1 = ws[OFF_PS1 + j]*(1.0f/N1F) - ws[OFF_V1B + j];
      float m2 = ws[OFF_PS2 + j]*(1.0f/N1F) - ws[OFF_V2B + j];
      pen = fmaf(m1, m1, pen);
      pen = fmaf(m2, m2, pen);
    }
    pen *= (1.0f/512.0f);
    float loss = ws[OFF_TERM]*(1.0f/N1F) + LAMc*pen;
    out[0] = loss;
    out[1] = pen;
  }
}

extern "C" void kernel_launch(void* const* d_in, const int* in_sizes, int n_in,
                              void* d_out, int out_size, void* d_ws, size_t ws_size,
                              hipStream_t stream){
  const float* dW = (const float*)d_in[0];
  // params flattened in sorted-key order: nn1..nn6, each (W0,b0,W1,b1,W2,b2)
  auto P = [&](int mlp, int layer, int wb) -> const float* {
    return (const float*)d_in[1 + mlp*6 + layer*2 + wb];
  };
  Mlp n1{P(0,0,0),P(0,0,1),P(0,1,0),P(0,1,1),P(0,2,0),P(0,2,1)};
  Mlp n2{P(1,0,0),P(1,0,1),P(1,1,0),P(1,1,1),P(1,2,0),P(1,2,1)};
  Mlp n3{P(2,0,0),P(2,0,1),P(2,1,0),P(2,1,1),P(2,2,0),P(2,2,1)};
  Mlp n4{P(3,0,0),P(3,0,1),P(3,1,0),P(3,1,1),P(3,2,0),P(3,2,1)};
  Mlp n5{P(4,0,0),P(4,0,1),P(4,1,0),P(4,1,1),P(4,2,0),P(4,2,1)};
  Mlp n6{P(5,0,0),P(5,0,1),P(5,1,0),P(5,1,1),P(5,2,0),P(5,2,1)};
  float* ws = (float*)d_ws;

  zero_kernel<<<4, 256, 0, stream>>>(ws);
  precompute_kernel<<<1, 64, 0, stream>>>(n1, n2, n5, n6, ws);
  sde_kernel<<<128, 128, 0, stream>>>(dW, n3, n4, ws);
  finalize_kernel<<<1, 32, 0, stream>>>(ws, (float*)d_out);
}
